// AttentionPairBias_927712936258
// MI455X (gfx1250) — matrix operand order repaired and checked by reading, not compiled, not run on hardware
//
#include <hip/hip_runtime.h>
#include <hip/hip_bf16.h>

#define HEADS 16
#define DH    64
#define DS_   384
#define DP_   128
#define N_    1024
#define DI    1024   // HEADS*DH

typedef _Float16 v16h __attribute__((ext_vector_type(16)));
typedef _Float16 h2   __attribute__((ext_vector_type(2)));
typedef float    v8f  __attribute__((ext_vector_type(8)));

static __device__ __forceinline__ v8f wmma_f16(v16h a, v16h b, v8f c) {
  // D = A(16x32 f16) x B(32x16 f16) + C(16x16 f32)
  return __builtin_amdgcn_wmma_f32_16x16x32_f16(false, a, false, b, (short)0, c,
                                                false, false);
}

static __device__ __forceinline__ h2 ldh2(const _Float16* p) {
  return *(const h2*)p;
}

// ---------------------------------------------------------------- prep kernels
__global__ void k_cvt(const float* __restrict__ src, _Float16* __restrict__ dst,
                      int count) {
  int i = blockIdx.x * blockDim.x + threadIdx.x;
  if (i < count) dst[i] = (_Float16)src[i];
}

// W is K x N row-major f32 -> WT is N x K row-major f16
__global__ void k_transpose(const float* __restrict__ W,
                            _Float16* __restrict__ WT, int K, int Nn) {
  int i = blockIdx.x * blockDim.x + threadIdx.x;
  if (i >= K * Nn) return;
  int k = i / Nn, n = i - k * Nn;
  WT[(size_t)n * K + k] = (_Float16)W[i];
}

// ------------------------------------------------- fused Q/K/V/G projections
// X(1024x384) @ W(384x1024); blockIdx.y selects matrix; 4 waves per block,
// one 16x16 output tile per wave.
__global__ __launch_bounds__(128) void k_qkvg(
    const _Float16* __restrict__ Xh,
    const _Float16* __restrict__ WqT, const _Float16* __restrict__ WkT,
    const _Float16* __restrict__ WvT, const _Float16* __restrict__ WgT,
    const float* __restrict__ bq,
    _Float16* __restrict__ Qh, _Float16* __restrict__ Kh,
    _Float16* __restrict__ Vt, float* __restrict__ G) {
  const int lane = threadIdx.x & 31;
  const int wav  = threadIdx.x >> 5;
  const int mat  = blockIdx.y;
  const int tile = blockIdx.x * 4 + wav;
  const int tm   = tile >> 6;   // 64 tiles per row of N
  const int tn   = tile & 63;
  const int hb   = lane >> 4;   // lane half
  const int lm   = lane & 15;

  const _Float16* WT = (mat == 0) ? WqT : (mat == 1) ? WkT : (mat == 2) ? WvT : WgT;

  v8f c = {};
  for (int kk = 0; kk < DS_; kk += 32) {
    v16h a, b;
#pragma unroll
    for (int v = 0; v < 8; ++v) {
      // A layout: lane holds row m=lm; vgpr v -> k = (v<4?0:16)+hb*8+(v&3)*2
      int ka = kk + ((v < 4) ? 0 : 16) + hb * 8 + (v & 3) * 2;
      h2 pa = ldh2(&Xh[(size_t)(tm * 16 + lm) * DS_ + ka]);
      a[2 * v] = pa.x; a[2 * v + 1] = pa.y;
      // B layout: lane holds col n=lm; vgpr v -> k = (v<4?0:16)+hb*8+(v&3)*2
      int kb = kk + ((v < 4) ? 0 : 16) + hb * 8 + (v & 3) * 2;
      h2 pb = ldh2(&WT[(size_t)(tn * 16 + lm) * DS_ + kb]);
      b[2 * v] = pb.x; b[2 * v + 1] = pb.y;
    }
    c = wmma_f16(a, b, c);
  }
#pragma unroll
  for (int r = 0; r < 8; ++r) {
    int m = tm * 16 + r + hb * 8;   // token
    int n = tn * 16 + lm;           // di index = h*64+d
    float val = c[r];
    if (mat == 0) {
      Qh[(size_t)m * DI + n] = (_Float16)(val + bq[n]);
    } else if (mat == 1) {
      Kh[(size_t)m * DI + n] = (_Float16)val;
    } else if (mat == 2) {
      Vt[(size_t)n * N_ + m] = (_Float16)val;       // Vt[(h*64+d)][token]
    } else {
      G[(size_t)m * DI + n] = 1.0f / (1.0f + __expf(-val));
    }
  }
}

// -------------------------------------- fused LN + pair-bias + flash attention
// grid.x = 64 query blocks of 16 rows; 16 waves = 16 heads per workgroup.
__global__ __launch_bounds__(512) void k_flash(
    const _Float16* __restrict__ Qh, const _Float16* __restrict__ Kh,
    const _Float16* __restrict__ Vt, const float* __restrict__ G,
    const float* __restrict__ pair, const float* __restrict__ gamma,
    const float* __restrict__ beta, const float* __restrict__ Wb,
    _Float16* __restrict__ OG) {
  __shared__ float    bias_s[16][32][16];   // [i_local][j_local][head] 32KB
  __shared__ _Float16 p_s[16][16][32];      // per-wave P staging        16KB

  const int lane = threadIdx.x & 31;
  const int hw   = threadIdx.x >> 5;   // wave id == head; also i_local for bias
  const int hb   = lane >> 4;
  const int lm   = lane & 15;
  const int i0   = blockIdx.x * 16;

  // LN folding: bias[p,h] = rstd*(x . (gamma*W))_h - rstd*mu*c1[h] + c2[h]
  float c1 = 0.f, c2 = 0.f;
  for (int d = 0; d < DP_; ++d) {
    float wv = Wb[d * HEADS + lm];
    c1 += gamma[d] * wv;
    c2 += beta[d] * wv;
  }
  v16h bwb[4];   // B-regs of (gamma .* W_bias), K=128 in 4 steps
#pragma unroll
  for (int s = 0; s < 4; ++s)
#pragma unroll
    for (int v = 0; v < 8; ++v) {
      int k = 32 * s + ((v < 4) ? 0 : 16) + hb * 8 + (v & 3) * 2;
      bwb[s][2 * v]     = (_Float16)(gamma[k] * Wb[k * HEADS + lm]);
      bwb[s][2 * v + 1] = (_Float16)(gamma[k + 1] * Wb[(k + 1) * HEADS + lm]);
    }

  // preload Q A-regs (16 rows x 64 d, two K=32 steps)
  v16h aq[2];
#pragma unroll
  for (int s = 0; s < 2; ++s)
#pragma unroll
    for (int v = 0; v < 8; ++v) {
      int d = 32 * s + ((v < 4) ? 0 : 16) + hb * 8 + (v & 3) * 2;
      h2 p = ldh2(&Qh[(size_t)(i0 + lm) * DI + hw * DH + d]);
      aq[s][2 * v] = p.x; aq[s][2 * v + 1] = p.y;
    }

  float Ml[8], Ll[8];
  v8f O[4] = {};
#pragma unroll
  for (int r = 0; r < 8; ++r) { Ml[r] = -3.0e38f; Ll[r] = 0.f; }

  for (int jb = 0; jb < N_; jb += 32) {
    if (jb + 32 < N_)
      __builtin_prefetch(pair + ((size_t)(i0 + hw) * N_ + (jb + 32 + lm)) * DP_, 0, 0);

    // ---- pair bias for rows (i0+hw, jb..jb+31): two 16-pair WMMA units ----
    for (int u = 0; u < 2; ++u) {
      int j = jb + u * 16 + lm;
      const float* pb = pair + ((size_t)(i0 + hw) * N_ + j) * DP_;
      float f[64];
      float sum = 0.f, sq = 0.f;
#pragma unroll
      for (int t = 0; t < 8; ++t) {
        int chunk = 2 * t + hb;   // lane half owns even/odd 8-feature chunks
        float4 x0 = *(const float4*)(pb + chunk * 8);
        float4 x1 = *(const float4*)(pb + chunk * 8 + 4);
        f[t * 8 + 0] = x0.x; f[t * 8 + 1] = x0.y; f[t * 8 + 2] = x0.z; f[t * 8 + 3] = x0.w;
        f[t * 8 + 4] = x1.x; f[t * 8 + 5] = x1.y; f[t * 8 + 6] = x1.z; f[t * 8 + 7] = x1.w;
        sum += x0.x + x0.y + x0.z + x0.w + x1.x + x1.y + x1.z + x1.w;
        sq  += x0.x * x0.x + x0.y * x0.y + x0.z * x0.z + x0.w * x0.w +
               x1.x * x1.x + x1.y * x1.y + x1.z * x1.z + x1.w * x1.w;
      }
      sum += __shfl_xor(sum, 16, 32);
      sq  += __shfl_xor(sq, 16, 32);
      float mu   = sum * (1.0f / DP_);
      float var  = sq * (1.0f / DP_) - mu * mu;
      float rstd = rsqrtf(var + 1e-5f);

      v8f cb = {};
#pragma unroll
      for (int s = 0; s < 4; ++s) {
        v16h a;
#pragma unroll
        for (int v = 0; v < 8; ++v) {
          int t = 2 * s + ((v < 4) ? 0 : 1);
          int o = (v & 3) * 2;
          a[2 * v]     = (_Float16)f[t * 8 + o];
          a[2 * v + 1] = (_Float16)f[t * 8 + o + 1];
        }
        cb = wmma_f16(a, bwb[s], cb);
      }
#pragma unroll
      for (int r = 0; r < 8; ++r) {
        int pm = r + hb * 8;                 // pair (j_local within unit)
        float rs = __shfl(rstd, pm, 32);
        float mm = __shfl(mu, pm, 32);
        bias_s[hw][u * 16 + pm][lm] = rs * cb[r] + (c2 - rs * mm * c1);
      }
    }
    __syncthreads();

    // ---- scores: S = Q K^T * scale + bias ----
    float s0[8], s1[8];
#pragma unroll
    for (int jt = 0; jt < 2; ++jt) {
      v8f sc = {};
#pragma unroll
      for (int s = 0; s < 2; ++s) {
        v16h b;
#pragma unroll
        for (int v = 0; v < 8; ++v) {
          int d = 32 * s + ((v < 4) ? 0 : 16) + hb * 8 + (v & 3) * 2;
          int j = jb + jt * 16 + lm;
          h2 p = ldh2(&Kh[(size_t)j * DI + hw * DH + d]);
          b[2 * v] = p.x; b[2 * v + 1] = p.y;
        }
        sc = wmma_f16(aq[s], b, sc);
      }
      float* dst = jt ? s1 : s0;
#pragma unroll
      for (int r = 0; r < 8; ++r) {
        int il = r + hb * 8;
        dst[r] = sc[r] * 0.125f + bias_s[il][jt * 16 + lm][hw];
      }
    }

    // ---- online softmax over the 32 new columns ----
    float alpha[8];
#pragma unroll
    for (int r = 0; r < 8; ++r) {
      float mx = fmaxf(s0[r], s1[r]);
#pragma unroll
      for (int msk = 1; msk < 16; msk <<= 1)
        mx = fmaxf(mx, __shfl_xor(mx, msk, 32));
      float nm = fmaxf(Ml[r], mx);
      alpha[r] = __expf(Ml[r] - nm);
      Ml[r] = nm;
      s0[r] = __expf(s0[r] - nm);
      s1[r] = __expf(s1[r] - nm);
      float rs = s0[r] + s1[r];
#pragma unroll
      for (int msk = 1; msk < 16; msk <<= 1)
        rs += __shfl_xor(rs, msk, 32);
      Ll[r] = Ll[r] * alpha[r] + rs;
    }
#pragma unroll
    for (int cc = 0; cc < 4; ++cc)
#pragma unroll
      for (int r = 0; r < 8; ++r) O[cc][r] *= alpha[r];

    // ---- stage P (C-layout -> row-major LDS -> A-layout) ----
#pragma unroll
    for (int r = 0; r < 8; ++r) {
      int il = r + hb * 8;
      p_s[hw][il][lm]      = (_Float16)s0[r];
      p_s[hw][il][16 + lm] = (_Float16)s1[r];
    }
    __builtin_amdgcn_wave_barrier();
    asm volatile("s_wait_dscnt 0" ::: "memory");
    __builtin_amdgcn_wave_barrier();

    v16h ap;
#pragma unroll
    for (int v = 0; v < 8; ++v) {
      int jl = ((v < 4) ? 0 : 16) + hb * 8 + (v & 3) * 2;
      h2 p = ldh2(&p_s[hw][lm][jl]);
      ap[2 * v] = p.x; ap[2 * v + 1] = p.y;
    }
    // ---- O += P @ V  (four 16-wide d chunks) ----
#pragma unroll
    for (int cc = 0; cc < 4; ++cc) {
      v16h b;
#pragma unroll
      for (int v = 0; v < 8; ++v) {
        int jl = ((v < 4) ? 0 : 16) + hb * 8 + (v & 3) * 2;
        int d  = cc * 16 + lm;
        h2 p = ldh2(&Vt[(size_t)(hw * DH + d) * N_ + jb + jl]);
        b[2 * v] = p.x; b[2 * v + 1] = p.y;
      }
      O[cc] = wmma_f16(ap, b, O[cc]);
    }
    __syncthreads();
  }

  // ---- finalize: /L, gate, store f16 ----
#pragma unroll
  for (int cc = 0; cc < 4; ++cc)
#pragma unroll
    for (int r = 0; r < 8; ++r) {
      int i  = i0 + r + hb * 8;
      int dg = hw * DH + cc * 16 + lm;
      float val = O[cc][r] / Ll[r];
      float g = G[(size_t)i * DI + dg];
      OG[(size_t)i * DI + dg] = (_Float16)(val * g);
    }
}

// ------------------------------------------------------ output projection GEMM
// Y(1024x384) = OG(1024x1024 f16) @ Wo; WoT is [384][1024] f16.
__global__ __launch_bounds__(128) void k_out(const _Float16* __restrict__ OG,
                                             const _Float16* __restrict__ WoT,
                                             float* __restrict__ out) {
  const int lane = threadIdx.x & 31;
  const int wav  = threadIdx.x >> 5;
  const int tile = blockIdx.x * 4 + wav;
  const int tm   = tile / 24;
  const int tn   = tile % 24;
  const int hb   = lane >> 4;
  const int lm   = lane & 15;
  v8f c = {};
  for (int kk = 0; kk < DI; kk += 32) {
    v16h a, b;
#pragma unroll
    for (int v = 0; v < 8; ++v) {
      int ka = kk + ((v < 4) ? 0 : 16) + hb * 8 + (v & 3) * 2;
      h2 pa = ldh2(&OG[(size_t)(tm * 16 + lm) * DI + ka]);
      a[2 * v] = pa.x; a[2 * v + 1] = pa.y;
      int kb = kk + ((v < 4) ? 0 : 16) + hb * 8 + (v & 3) * 2;
      h2 pb = ldh2(&WoT[(size_t)(tn * 16 + lm) * DI + kb]);
      b[2 * v] = pb.x; b[2 * v + 1] = pb.y;
    }
    c = wmma_f16(a, b, c);
  }
#pragma unroll
  for (int r = 0; r < 8; ++r)
    out[(size_t)(tm * 16 + r + hb * 8) * DS_ + tn * 16 + lm] = c[r];
}

// ---------------------------------------------------------------------- launch
extern "C" void kernel_launch(void* const* d_in, const int* in_sizes, int n_in,
                              void* d_out, int out_size, void* d_ws, size_t ws_size,
                              hipStream_t stream) {
  const float* single = (const float*)d_in[0];
  const float* pair   = (const float*)d_in[1];
  const float* gamma  = (const float*)d_in[2];
  const float* beta   = (const float*)d_in[3];
  const float* Wb     = (const float*)d_in[4];
  const float* Wq     = (const float*)d_in[5];
  const float* bq     = (const float*)d_in[6];
  const float* Wk     = (const float*)d_in[7];
  const float* Wv     = (const float*)d_in[8];
  const float* Wg     = (const float*)d_in[9];
  const float* Wo     = (const float*)d_in[10];
  float* out = (float*)d_out;

  char* ws = (char*)d_ws;
  _Float16* Xh  = (_Float16*)(ws + 0);          // 1024*384*2
  _Float16* WqT = (_Float16*)(ws + 786432);
  _Float16* WkT = (_Float16*)(ws + 1572864);
  _Float16* WvT = (_Float16*)(ws + 2359296);
  _Float16* WgT = (_Float16*)(ws + 3145728);
  _Float16* WoT = (_Float16*)(ws + 3932160);    // 384*1024*2
  _Float16* Qh  = (_Float16*)(ws + 4718592);    // 1024*1024*2
  _Float16* Kh  = (_Float16*)(ws + 6815744);
  _Float16* Vt  = (_Float16*)(ws + 8912896);    // [h*64+d][token]
  float*    G   = (float*)   (ws + 11010048);   // 1024*1024*4
  _Float16* OG  = (_Float16*)(ws + 15204352);   // 1024*1024*2  (end 17301504)

  k_cvt<<<1536, 256, 0, stream>>>(single, Xh, 393216);
  k_transpose<<<1536, 256, 0, stream>>>(Wq, WqT, 384, 1024);
  k_transpose<<<1536, 256, 0, stream>>>(Wk, WkT, 384, 1024);
  k_transpose<<<1536, 256, 0, stream>>>(Wv, WvT, 384, 1024);
  k_transpose<<<1536, 256, 0, stream>>>(Wg, WgT, 384, 1024);
  k_transpose<<<1536, 256, 0, stream>>>(Wo, WoT, 1024, 384);
  k_qkvg<<<dim3(1024, 4), 128, 0, stream>>>(Xh, WqT, WkT, WvT, WgT, bq,
                                            Qh, Kh, Vt, G);
  k_flash<<<64, 512, 0, stream>>>(Qh, Kh, Vt, G, pair, gamma, beta, Wb, OG);
  k_out<<<384, 128, 0, stream>>>(OG, WoT, out);
}
